// HiTMicroTiny_81063212745164
// MI455X (gfx1250) — matrix operand order repaired and checked by reading, not compiled, not run on hardware
//
#include <hip/hip_runtime.h>
#include <hip/hip_bf16.h>
#include <math.h>

// ---------------------------------------------------------------------------
// ViT-tiny forward for MI455X (gfx1250, wave32).
// All GEMMs + attention use v_wmma_f32_16x16x32_bf16 (f32 accumulate).
// GEMM: 32x64 output tile per wave (2 A frags x 4 B frags, 8 accumulators)
// to raise L2 arithmetic intensity to ~21 FLOP/byte.
// ---------------------------------------------------------------------------

typedef __bf16 bf16;
typedef __attribute__((ext_vector_type(16))) __bf16 v16bf;
typedef __attribute__((ext_vector_type(8)))  __bf16 v8bf;
typedef __attribute__((ext_vector_type(8)))  float  v8f;

#define DEPTH   12
#define DMODEL  192
#define NHEADS  3
#define HDIM    64
#define MCROPS  85
#define GRIDH   14
#define NTOK    282      // 1 cls + 281 patches
#define NPAD    288      // 18 * 16
#define BATCH   128
#define MROWS   (BATCH*NTOK)   // 36096 (= 1128 * 32)
#define EROWS   (BATCH*281)    // 35968 (= 1124 * 32)

static __device__ __forceinline__ v8f wmma_bf16(v16bf a, v16bf b, v8f c) {
    return __builtin_amdgcn_wmma_f32_16x16x32_bf16(false, a, false, b,
                                                   (short)0, c, false, false);
}

// A fragment (16x32 bf16, row-major source, contiguous K).
// lane L: row m = m0 + (L&15), half = L>>4; elem i -> k = half*16 + i
static __device__ __forceinline__ v16bf load_a_frag(const bf16* rowp /* = &A[m][k0+half*16] */) {
    v8bf lo = *(const v8bf*)(rowp);
    v8bf hi = *(const v8bf*)(rowp + 8);
    v16bf a;
#pragma unroll
    for (int i = 0; i < 8; ++i) { a[i] = lo[i]; a[8 + i] = hi[i]; }
    return a;
}

// ---------------------------------------------------------------------------
// Generic WMMA GEMM: out[m][n] = sum_k A[m][k] * W[n][k]   (A: M x K, W: N x K)
// One wave per 32x64 output tile. MODE selects fused epilogue.
//   MODE 0: patch embed -> x f32 with +bias +PE (rows are b*281+p, token p+1)
//   MODE 1: qkv -> scatter bf16 into q[(b,h),t,d], k[(b,h),t,d], vT[(b,h),d,t] (+bias)
//   MODE 2: residual: x[m][n] += acc + bias  (N == 192)
//   MODE 3: fc1: outb[m][n] = bf16(gelu(acc + bias))
// ---------------------------------------------------------------------------
template <int MODE>
__global__ __launch_bounds__(32)
void gemm_wmma(const bf16* __restrict__ A, const bf16* __restrict__ W,
               const float* __restrict__ bias, int Mrows, int K, int N,
               float* __restrict__ outf, bf16* __restrict__ outb,
               const float* __restrict__ pe,
               bf16* __restrict__ qb, bf16* __restrict__ kb, bf16* __restrict__ vb)
{
    const int lane = threadIdx.x & 31;
    const int nl   = lane & 15;
    const int half = lane >> 4;
    const int m0   = blockIdx.x * 32;
    const int n0   = blockIdx.y * 64;

    v8f acc[2][4] = {{v8f{}, v8f{}, v8f{}, v8f{}},
                     {v8f{}, v8f{}, v8f{}, v8f{}}};

    const bf16* arow0 = A + (size_t)(m0 + nl) * K + half * 16;
    const bf16* arow1 = arow0 + (size_t)16 * K;
    for (int k0 = 0; k0 < K; k0 += 32) {
        v16bf a0 = load_a_frag(arow0 + k0);
        v16bf a1 = load_a_frag(arow1 + k0);
#pragma unroll
        for (int g = 0; g < 4; ++g) {
            const bf16* wrow = W + (size_t)(n0 + g * 16 + nl) * K + k0 + half * 16;
            v16bf b = *(const v16bf*)wrow;
            acc[0][g] = wmma_bf16(a0, b, acc[0][g]);
            acc[1][g] = wmma_bf16(a1, b, acc[1][g]);
        }
    }

    // C layout: vgpr r, lane -> (m = m0 + s*16 + r + 8*half, n = n0 + g*16 + nl)
#pragma unroll
    for (int s = 0; s < 2; ++s) {
#pragma unroll
        for (int g = 0; g < 4; ++g) {
            const int n  = n0 + g * 16 + nl;
            const float bv = bias ? bias[n] : 0.0f;
#pragma unroll
            for (int r = 0; r < 8; ++r) {
                const int m = m0 + s * 16 + r + 8 * half;
                if (m >= Mrows) continue;
                float v = acc[s][g][r] + bv;
                if (MODE == 0) {
                    int b_ = m / 281, p = m % 281, t = p + 1;
                    outf[((size_t)b_ * NTOK + t) * DMODEL + n] = v + pe[t * DMODEL + n];
                } else if (MODE == 1) {
                    int b_ = m / NTOK, t = m % NTOK;
                    int part = n / DMODEL, rem = n % DMODEL;
                    int h = rem / HDIM, d = rem % HDIM;
                    size_t bh = (size_t)b_ * NHEADS + h;
                    bf16 bvv = (bf16)v;
                    if (part == 0)      qb[(bh * NPAD + t) * HDIM + d] = bvv;
                    else if (part == 1) kb[(bh * NPAD + t) * HDIM + d] = bvv;
                    else                vb[(bh * HDIM + d) * NPAD + t] = bvv;
                } else if (MODE == 2) {
                    outf[(size_t)m * DMODEL + n] += v;
                } else {
                    float gv = 0.5f * v * (1.0f + erff(v * 0.70710678118654752f));
                    outb[(size_t)m * 768 + n] = (bf16)gv;
                }
            }
        }
    }
}

// ---------------------------------------------------------------------------
// Flash attention: one wave per (b*NH+h, 16-row tile). hd = 64, N padded to 288.
// ---------------------------------------------------------------------------
__global__ __launch_bounds__(32)
void flash_attn(const bf16* __restrict__ qb, const bf16* __restrict__ kb,
                const bf16* __restrict__ vb, bf16* __restrict__ ob)
{
    __shared__ __align__(64) bf16 plds[16 * 32];

    const int bh   = blockIdx.x;          // b*NH + h
    const int tile = blockIdx.y;          // 0..17
    const int lane = threadIdx.x & 31;
    const int nl   = lane & 15;
    const int half = lane >> 4;
    const int m0   = tile * 16;

    // Q fragments for this row tile (K = 64 -> two k=32 chunks), kept in regs.
    v16bf qa[2];
    {
        const bf16* r = qb + ((size_t)bh * NPAD + m0 + nl) * HDIM + half * 16;
        qa[0] = load_a_frag(r);
        qa[1] = load_a_frag(r + 32);
    }

    float mstate[8], lstate[8];
#pragma unroll
    for (int r = 0; r < 8; ++r) { mstate[r] = -1e30f; lstate[r] = 0.0f; }
    v8f o[4] = {v8f{}, v8f{}, v8f{}, v8f{}};
    const float scale = 0.125f; // 64^-0.5

    for (int jj = 0; jj < 9; ++jj) {
        const int j0 = jj * 32;
        v8f s[2];
#pragma unroll
        for (int sub = 0; sub < 2; ++sub) {
            // B frag of K^T: lane -> column token j0+sub*16+nl; elems over d
            const bf16* kr = kb + ((size_t)bh * NPAD + j0 + sub * 16 + nl) * HDIM + half * 16;
            v8f c = {};
            c = wmma_bf16(qa[0], *(const v16bf*)(kr),      c);
            c = wmma_bf16(qa[1], *(const v16bf*)(kr + 32), c);
            const int tok = j0 + sub * 16 + nl;
#pragma unroll
            for (int r = 0; r < 8; ++r)
                c[r] = (tok < NTOK) ? c[r] * scale : -1e30f;
            s[sub] = c;
        }
        // Online softmax: reduce over n (lanes within each 16-lane half).
#pragma unroll
        for (int r = 0; r < 8; ++r) {
            float mx = fmaxf(s[0][r], s[1][r]);
#pragma unroll
            for (int msk = 8; msk >= 1; msk >>= 1) mx = fmaxf(mx, __shfl_xor(mx, msk, 32));
            const float nmax = fmaxf(mstate[r], mx);
            const float e0 = __expf(s[0][r] - nmax);
            const float e1 = __expf(s[1][r] - nmax);
            s[0][r] = e0; s[1][r] = e1;
            float ps = e0 + e1;
#pragma unroll
            for (int msk = 8; msk >= 1; msk >>= 1) ps += __shfl_xor(ps, msk, 32);
            const float corr = __expf(mstate[r] - nmax);
            lstate[r] = lstate[r] * corr + ps;
            mstate[r] = nmax;
#pragma unroll
            for (int g = 0; g < 4; ++g) o[g][r] *= corr;
        }
        // Re-layout P (16x32, C layout) -> A fragment through LDS.
#pragma unroll
        for (int sub = 0; sub < 2; ++sub)
#pragma unroll
            for (int r = 0; r < 8; ++r)
                plds[(r + 8 * half) * 32 + sub * 16 + nl] = (bf16)s[sub][r];
        asm volatile("s_wait_dscnt 0" ::: "memory");
        v16bf pa = load_a_frag(plds + nl * 32 + half * 16);
        asm volatile("s_wait_dscnt 0" ::: "memory");
        // o += P @ V : B frag from transposed V, contiguous over tokens.
#pragma unroll
        for (int g = 0; g < 4; ++g) {
            const bf16* vr = vb + ((size_t)bh * HDIM + g * 16 + nl) * NPAD + j0 + half * 16;
            o[g] = wmma_bf16(pa, *(const v16bf*)vr, o[g]);
        }
    }

    const int b_ = bh / NHEADS, h = bh % NHEADS;
#pragma unroll
    for (int r = 0; r < 8; ++r) {
        const int m = m0 + r + 8 * half;
        if (m >= NTOK) continue;
        const float invl = 1.0f / lstate[r];
#pragma unroll
        for (int g = 0; g < 4; ++g)
            ob[((size_t)b_ * NTOK + m) * DMODEL + h * HDIM + g * 16 + nl] =
                (bf16)(o[g][r] * invl);
    }
}

// ---------------------------------------------------------------------------
// LayerNorm: 1 wave per row of 192, writes bf16.
// ---------------------------------------------------------------------------
__global__ void ln_kernel(const float* __restrict__ x, const float* __restrict__ w,
                          const float* __restrict__ b, bf16* __restrict__ out, int Mrows)
{
    const int wid  = (int)((blockIdx.x * (size_t)blockDim.x + threadIdx.x) >> 5);
    if (wid >= Mrows) return;
    const int lane = threadIdx.x & 31;
    const float* row = x + (size_t)wid * DMODEL;
    float v[6], s = 0.0f;
#pragma unroll
    for (int i = 0; i < 6; ++i) { v[i] = row[lane + i * 32]; s += v[i]; }
#pragma unroll
    for (int m = 16; m >= 1; m >>= 1) s += __shfl_xor(s, m, 32);
    const float mean = s * (1.0f / DMODEL);
    float var = 0.0f;
#pragma unroll
    for (int i = 0; i < 6; ++i) { float d = v[i] - mean; var += d * d; }
#pragma unroll
    for (int m = 16; m >= 1; m >>= 1) var += __shfl_xor(var, m, 32);
    const float inv = rsqrtf(var * (1.0f / DMODEL) + 1e-6f);
#pragma unroll
    for (int i = 0; i < 6; ++i) {
        const int c = lane + i * 32;
        out[(size_t)wid * DMODEL + c] = (bf16)((v[i] - mean) * inv * w[c] + b[c]);
    }
}

// ---------------------------------------------------------------------------
// Micro-crop bilinear (8x8 -> 16x16) + fine 16x16 patch gather -> bf16 patches
// ---------------------------------------------------------------------------
__global__ void build_patches(const float* __restrict__ images, const int* __restrict__ top,
                              const int* __restrict__ left, bf16* __restrict__ patches)
{
    const size_t total = (size_t)BATCH * 281 * 768;
    size_t idx = (size_t)blockIdx.x * blockDim.x + threadIdx.x;
    if (idx >= total) return;
    const int feat = (int)(idx % 768);
    const size_t t = idx / 768;
    const int p = (int)(t % 281);
    const int b = (int)(t / 281);
    const int c = feat >> 8, py = (feat >> 4) & 15, px = feat & 15;
    const float* img = images + ((size_t)b * 3 + c) * 224 * 224;
    float val;
    if (p < MCROPS) {
        float sy = fmaxf(py * 0.5f - 0.25f, 0.0f);
        int y0 = min((int)sy, 7), y1 = min(y0 + 1, 7);
        float wy1 = sy - (float)y0, wy0 = 1.0f - wy1;
        float sx = fmaxf(px * 0.5f - 0.25f, 0.0f);
        int x0 = min((int)sx, 7), x1 = min(x0 + 1, 7);
        float wx1 = sx - (float)x0, wx0 = 1.0f - wx1;
        const float* base = img + (size_t)top[p] * 224 + left[p];
        float v00 = base[y0 * 224 + x0], v01 = base[y0 * 224 + x1];
        float v10 = base[y1 * 224 + x0], v11 = base[y1 * 224 + x1];
        val = wy0 * (wx0 * v00 + wx1 * v01) + wy1 * (wx0 * v10 + wx1 * v11);
    } else {
        const int f = p - MCROPS, gy = f / GRIDH, gx = f % GRIDH;
        val = img[(size_t)(gy * 16 + py) * 224 + gx * 16 + px];
    }
    patches[idx] = (bf16)val;
}

// Fourier positional embedding table (282 x 192), row 0 unused (cls).
__global__ void build_pe(const int* __restrict__ top, const int* __restrict__ left,
                         float* __restrict__ pe)
{
    const int idx = blockIdx.x * blockDim.x + threadIdx.x;
    if (idx >= NTOK * DMODEL) return;
    const int t = idx / DMODEL, n = idx % DMODEL;
    if (t == 0) { pe[idx] = 0.0f; return; }
    const int p = t - 1;
    const int axis = n / 64, j = n % 64;
    float coord;
    if (p < MCROPS) {
        if (axis == 0)      coord = ((float)left[p] + 4.0f) / 224.0f;
        else if (axis == 1) coord = ((float)top[p]  + 4.0f) / 224.0f;
        else                coord = 8.0f / 224.0f;
    } else {
        const int f = p - MCROPS, gy = f / GRIDH, gx = f % GRIDH;
        if (axis == 0)      coord = ((float)gx + 0.5f) / 14.0f;
        else if (axis == 1) coord = ((float)gy + 0.5f) / 14.0f;
        else                coord = 1.0f / 14.0f;
    }
    const float ang = coord * exp2f((float)(j & 31));
    pe[idx] = (j < 32) ? __sinf(ang) : __cosf(ang);
}

__global__ void init_cls(const float* __restrict__ cls_token, const float* __restrict__ cls_pos,
                         float* __restrict__ x)
{
    const int idx = blockIdx.x * blockDim.x + threadIdx.x;
    if (idx >= BATCH * DMODEL) return;
    const int b = idx / DMODEL, n = idx % DMODEL;
    x[(size_t)b * NTOK * DMODEL + n] = cls_token[n] + cls_pos[n];
}

__global__ void f32_to_bf16(const float* __restrict__ in, bf16* __restrict__ out, size_t n)
{
    size_t i = (size_t)blockIdx.x * blockDim.x + threadIdx.x;
    if (i < n) out[i] = (bf16)in[i];
}

// Final LN on cls token + 10-way head. One wave per batch element.
__global__ __launch_bounds__(32)
void head_kernel(const float* __restrict__ x, const float* __restrict__ nw,
                 const float* __restrict__ nb, const float* __restrict__ hw,
                 const float* __restrict__ hb, float* __restrict__ out)
{
    const int b = blockIdx.x;
    const int lane = threadIdx.x & 31;
    const float* row = x + (size_t)b * NTOK * DMODEL;
    float v[6], s = 0.0f;
#pragma unroll
    for (int i = 0; i < 6; ++i) { v[i] = row[lane + i * 32]; s += v[i]; }
#pragma unroll
    for (int m = 16; m >= 1; m >>= 1) s += __shfl_xor(s, m, 32);
    const float mean = s * (1.0f / DMODEL);
    float var = 0.0f;
#pragma unroll
    for (int i = 0; i < 6; ++i) { float d = v[i] - mean; var += d * d; }
#pragma unroll
    for (int m = 16; m >= 1; m >>= 1) var += __shfl_xor(var, m, 32);
    const float inv = rsqrtf(var * (1.0f / DMODEL) + 1e-6f);
    float h[6];
#pragma unroll
    for (int i = 0; i < 6; ++i) {
        const int c = lane + i * 32;
        h[i] = (v[i] - mean) * inv * nw[c] + nb[c];
    }
    for (int cls = 0; cls < 10; ++cls) {
        float acc = 0.0f;
#pragma unroll
        for (int i = 0; i < 6; ++i) acc += h[i] * hw[cls * DMODEL + lane + i * 32];
#pragma unroll
        for (int m = 16; m >= 1; m >>= 1) acc += __shfl_xor(acc, m, 32);
        if (lane == 0) out[b * 10 + cls] = acc + hb[cls];
    }
}

// ---------------------------------------------------------------------------
extern "C" void kernel_launch(void* const* d_in, const int* in_sizes, int n_in,
                              void* d_out, int out_size, void* d_ws, size_t ws_size,
                              hipStream_t stream)
{
    (void)in_sizes; (void)n_in; (void)out_size; (void)ws_size;
    const float* images   = (const float*)d_in[0];
    const int*   top      = (const int*)  d_in[1];
    const int*   left     = (const int*)  d_in[2];
    const float* patch_W  = (const float*)d_in[3];
    const float* patch_b  = (const float*)d_in[4];
    const float* cls_tok  = (const float*)d_in[5];
    const float* cls_pos  = (const float*)d_in[6];
    const float* ln1_w    = (const float*)d_in[7];
    const float* ln1_b    = (const float*)d_in[8];
    const float* qkv_w    = (const float*)d_in[9];
    const float* qkv_b    = (const float*)d_in[10];
    const float* proj_w   = (const float*)d_in[11];
    const float* proj_b   = (const float*)d_in[12];
    const float* ln2_w    = (const float*)d_in[13];
    const float* ln2_b    = (const float*)d_in[14];
    const float* fc1_w    = (const float*)d_in[15];
    const float* fc1_b    = (const float*)d_in[16];
    const float* fc2_w    = (const float*)d_in[17];
    const float* fc2_b    = (const float*)d_in[18];
    const float* norm_w   = (const float*)d_in[19];
    const float* norm_b   = (const float*)d_in[20];
    const float* head_w   = (const float*)d_in[21];
    const float* head_b   = (const float*)d_in[22];

    // ---- workspace layout (deterministic) ----
    char* wsb = (char*)d_ws;
    size_t off = 0;
    auto take = [&](size_t bytes) -> void* {
        void* p = wsb + off;
        off = (off + bytes + 255) & ~(size_t)255;
        return p;
    };
    const size_t nWp  = (size_t)DMODEL * 768;                 // patch_W
    const size_t nWq  = (size_t)DEPTH * 3 * DMODEL * DMODEL;  // qkv_w
    const size_t nWpr = (size_t)DEPTH * DMODEL * DMODEL;      // proj_w
    const size_t nWf1 = (size_t)DEPTH * 4 * DMODEL * DMODEL;  // fc1_w
    const size_t nWf2 = (size_t)DEPTH * DMODEL * 4 * DMODEL;  // fc2_w
    bf16*  wPatch = (bf16*)take(nWp  * 2);
    bf16*  wQkv   = (bf16*)take(nWq  * 2);
    bf16*  wProj  = (bf16*)take(nWpr * 2);
    bf16*  wFc1   = (bf16*)take(nWf1 * 2);
    bf16*  wFc2   = (bf16*)take(nWf2 * 2);
    float* pe     = (float*)take((size_t)NTOK * DMODEL * 4);
    float* x      = (float*)take((size_t)MROWS * DMODEL * 4);
    bf16*  hbuf   = (bf16*) take((size_t)MROWS * DMODEL * 2);       // LN out / attn out
    bf16*  big    = (bf16*) take((size_t)MROWS * 768 * 2);          // patches & fc1 act
    const size_t nQKV = (size_t)BATCH * NHEADS * NPAD * HDIM;
    bf16*  qbuf   = (bf16*)take(nQKV * 2);
    bf16*  kbuf   = (bf16*)take(nQKV * 2);
    bf16*  vbuf   = (bf16*)take(nQKV * 2);

    // zero q/k/v once so padded token rows (282..287) are benign
    hipMemsetAsync(qbuf, 0, nQKV * 2, stream);
    hipMemsetAsync(kbuf, 0, nQKV * 2, stream);
    hipMemsetAsync(vbuf, 0, nQKV * 2, stream);

    // ---- weight conversion to bf16 ----
    auto conv = [&](const float* in, bf16* o, size_t n) {
        f32_to_bf16<<<(unsigned)((n + 255) / 256), 256, 0, stream>>>(in, o, n);
    };
    conv(patch_W, wPatch, nWp);
    conv(qkv_w,  wQkv,  nWq);
    conv(proj_w, wProj, nWpr);
    conv(fc1_w,  wFc1,  nWf1);
    conv(fc2_w,  wFc2,  nWf2);

    // ---- tokenization + PE + cls ----
    {
        const size_t tot = (size_t)BATCH * 281 * 768;
        build_patches<<<(unsigned)((tot + 255) / 256), 256, 0, stream>>>(images, top, left, big);
        build_pe<<<(NTOK * DMODEL + 255) / 256, 256, 0, stream>>>(top, left, pe);
        init_cls<<<(BATCH * DMODEL + 255) / 256, 256, 0, stream>>>(cls_tok, cls_pos, x);
    }

    // ---- patch embed: (B*281,768) @ (192,768)^T + bias + PE -> x ----
    gemm_wmma<0><<<dim3(EROWS / 32, DMODEL / 64), 32, 0, stream>>>(
        big, wPatch, patch_b, EROWS, 768, DMODEL, x, nullptr, pe,
        nullptr, nullptr, nullptr);

    const int mtiles = MROWS / 32;           // 1128
    const unsigned ln_grid = (MROWS * 32 + 255) / 256;

    for (int i = 0; i < DEPTH; ++i) {
        // LN1 -> hbuf (bf16)
        ln_kernel<<<ln_grid, 256, 0, stream>>>(x, ln1_w + i * DMODEL, ln1_b + i * DMODEL,
                                               hbuf, MROWS);
        // QKV: (M,192) @ (576,192)^T -> q/k/vT bf16
        gemm_wmma<1><<<dim3(mtiles, (3 * DMODEL) / 64), 32, 0, stream>>>(
            hbuf, wQkv + (size_t)i * 3 * DMODEL * DMODEL, qkv_b + i * 3 * DMODEL,
            MROWS, DMODEL, 3 * DMODEL, nullptr, nullptr, nullptr, qbuf, kbuf, vbuf);
        // Flash attention -> hbuf (attn output, bf16)
        flash_attn<<<dim3(BATCH * NHEADS, NPAD / 16), 32, 0, stream>>>(qbuf, kbuf, vbuf, hbuf);
        // proj + residual -> x
        gemm_wmma<2><<<dim3(mtiles, DMODEL / 64), 32, 0, stream>>>(
            hbuf, wProj + (size_t)i * DMODEL * DMODEL, proj_b + i * DMODEL,
            MROWS, DMODEL, DMODEL, x, nullptr, nullptr, nullptr, nullptr, nullptr);
        // LN2 -> hbuf
        ln_kernel<<<ln_grid, 256, 0, stream>>>(x, ln2_w + i * DMODEL, ln2_b + i * DMODEL,
                                               hbuf, MROWS);
        // fc1 + GELU -> big (bf16)
        gemm_wmma<3><<<dim3(mtiles, 768 / 64), 32, 0, stream>>>(
            hbuf, wFc1 + (size_t)i * 4 * DMODEL * DMODEL, fc1_b + i * 4 * DMODEL,
            MROWS, DMODEL, 768, nullptr, big, nullptr, nullptr, nullptr, nullptr);
        // fc2 + residual -> x
        gemm_wmma<2><<<dim3(mtiles, DMODEL / 64), 32, 0, stream>>>(
            big, wFc2 + (size_t)i * DMODEL * 4 * DMODEL, fc2_b + i * DMODEL,
            MROWS, 768, DMODEL, x, nullptr, nullptr, nullptr, nullptr, nullptr);
    }

    // ---- final LN + head on cls rows ----
    head_kernel<<<BATCH, 32, 0, stream>>>(x, norm_w, norm_b, head_w, head_b, (float*)d_out);
}
